// SelfAttention_86964497809841
// MI455X (gfx1250) — compile-verified, matrix operand order checked by reading, not run on hardware
//
#include <hip/hip_runtime.h>
#include <hip/hip_bf16.h>
#include <stdint.h>

typedef __attribute__((ext_vector_type(16))) __bf16 bf16x16;
typedef __attribute__((ext_vector_type(8)))  __bf16 bf16x8;
typedef __attribute__((ext_vector_type(8)))  float  f32x8;

// ---------------------------------------------------------------------------
// CDNA5 async global->LDS copy (ASYNCcnt-tracked), via inline asm.
// ---------------------------------------------------------------------------
__device__ __forceinline__ void async_load_b128(uint32_t lds_addr, const void* gptr) {
  asm volatile("global_load_async_to_lds_b128 %0, %1, off"
               :: "v"(lds_addr), "v"(gptr)
               : "memory");
}
__device__ __forceinline__ void wait_async0() {
  asm volatile("s_wait_asynccnt 0x0" ::: "memory");
}
__device__ __forceinline__ uint32_t lds_addr_of(const void* p) {
  return (uint32_t)(uintptr_t)p;   // generic LDS ptr: low 32 bits = LDS offset
}

// ---------------------------------------------------------------------------
// f32 -> bf16 convert
// ---------------------------------------------------------------------------
__global__ __launch_bounds__(256) void f32_to_bf16_kernel(
    const float* __restrict__ in, __bf16* __restrict__ out, int n) {
  int i = blockIdx.x * 256 + threadIdx.x;
  if (i < n) out[i] = (__bf16)in[i];
}

// ---------------------------------------------------------------------------
// WMMA fragment load from an LDS tile (16 rows x 32 K), row stride in elems.
// CDNA5 16-bit A/B operand layout:
//   lanes 0-15  : row = lane,    K = {0..7, 16..23}
//   lanes 16-31 : row = lane-16, K = {8..15, 24..31}
// ---------------------------------------------------------------------------
__device__ __forceinline__ bf16x16 load_frag(const __bf16* base, int stride_elems,
                                             int lane) {
  const int r = lane & 15;
  const int h = (lane >> 4) & 1;
  const __bf16* p = base + r * stride_elems + h * 8;
  bf16x8 lo = *(const bf16x8*)(p);
  bf16x8 hi = *(const bf16x8*)(p + 16);
  bf16x16 f;
#pragma unroll
  for (int i = 0; i < 8; ++i) { f[i] = lo[i]; f[i + 8] = hi[i]; }
  return f;
}

// ---------------------------------------------------------------------------
// Tiled WMMA GEMM:  C[m,n] = sum_k A[m,k] * Bt[n,k]
// A  : row-major [M, K];  Bt : row-major [N, K] (B consumed transposed)
// Block tile 256(M) x 128(N), 8 waves (4M x 2N), wave tile 64x64 =
// 4x4 WMMA 16x16x32 tiles (16 WMMA vs 16 ds_loads per K-step per wave).
// K-step 32. Double-buffered LDS fed by async global->LDS loads.
// MODE 0: QKV epilogue (de-interleave to q*scale / k / v, bf16)
// MODE 1: plain f32 store to C
// ---------------------------------------------------------------------------
#define BM 256
#define BN 128
#define BK 32
#define LDS_STR 40   // 32 + 16B pad; row stride 80B (multiple of 16 -> b128 ok)

template <int MODE>
__global__ __launch_bounds__(256) void gemm_wmma_kernel(
    const __bf16* __restrict__ A, long long aBatch, int lda,
    const __bf16* __restrict__ B, long long bBatch, int ldb,
    float* __restrict__ C, long long cBatch, int ldc,
    int K, float qscale,
    __bf16* __restrict__ qb, __bf16* __restrict__ kb, __bf16* __restrict__ vb) {
  __shared__ __bf16 As[2][BM * LDS_STR];   // 2 x 20 KB
  __shared__ __bf16 Bs[2][BN * LDS_STR];   // 2 x 10 KB

  const int tid  = threadIdx.x;
  const int lane = tid & 31;
  const int wave = tid >> 5;
  const int wm   = wave & 3;   // 0..3 along M (64 rows each)
  const int wn   = wave >> 2;  // 0..1 along N (64 cols each)
  const int m0   = blockIdx.y * BM;
  const int n0   = blockIdx.x * BN;
  const int z    = blockIdx.z;

  A += (long long)z * aBatch;
  B += (long long)z * bBatch;
  if (MODE == 1) C += (long long)z * cBatch;

  f32x8 acc[4][4];
#pragma unroll
  for (int i = 0; i < 4; ++i)
#pragma unroll
    for (int j = 0; j < 4; ++j)
#pragma unroll
      for (int e = 0; e < 8; ++e) acc[i][j][e] = 0.0f;

  const int chunk = tid & 3;   // 16B chunk within the 32-wide K slab
  const int row0  = tid >> 2;  // 0..63

  // One 256x32 A tile (4 passes) + 128x32 B tile (2 passes): 6 async loads/thread.
  auto issue_tile = [&](int buf, int k0) {
#pragma unroll
    for (int p = 0; p < 4; ++p) {
      const int r = row0 + p * 64;
      async_load_b128(lds_addr_of(&As[buf][r * LDS_STR + chunk * 8]),
                      &A[(long long)(m0 + r) * lda + k0 + chunk * 8]);
    }
#pragma unroll
    for (int p = 0; p < 2; ++p) {
      const int r = row0 + p * 64;
      async_load_b128(lds_addr_of(&Bs[buf][r * LDS_STR + chunk * 8]),
                      &B[(long long)(n0 + r) * ldb + k0 + chunk * 8]);
    }
  };

  issue_tile(0, 0);
  int buf = 0;

  for (int k0 = 0; k0 < K; k0 += BK) {
    wait_async0();        // this wave's writes into `buf` are in LDS
    __syncthreads();      // everyone's writes into `buf` are in LDS
    if (k0 + BK < K) issue_tile(buf ^ 1, k0 + BK);  // prefetch next tile

    bf16x16 af[4], bfv[4];
#pragma unroll
    for (int i = 0; i < 4; ++i)
      af[i] = load_frag(&As[buf][(wm * 64 + i * 16) * LDS_STR], LDS_STR, lane);
#pragma unroll
    for (int j = 0; j < 4; ++j)
      bfv[j] = load_frag(&Bs[buf][(wn * 64 + j * 16) * LDS_STR], LDS_STR, lane);

#pragma unroll
    for (int i = 0; i < 4; ++i)
#pragma unroll
      for (int j = 0; j < 4; ++j)
        acc[i][j] = __builtin_amdgcn_wmma_f32_16x16x32_bf16(
            false, af[i], false, bfv[j], (short)0, acc[i][j], false, false);

    buf ^= 1;
  }

  // Epilogue. C-fragment layout: VGPR e, lane l -> M = e + 8*(l>>4), N = l&15
  const int h     = lane >> 4;
  const int nlane = lane & 15;
#pragma unroll
  for (int j = 0; j < 4; ++j) {
    const int gn = n0 + wn * 64 + j * 16 + nlane;
    int which = 0, col = 0;
    if (MODE == 0) { which = gn % 3; col = gn / 3; }
#pragma unroll
    for (int i = 0; i < 4; ++i) {
#pragma unroll
      for (int e = 0; e < 8; ++e) {
        const int gm = m0 + wm * 64 + i * 16 + e + 8 * h;
        const float val = acc[i][j][e];
        if (MODE == 0) {
          const long long idx = (long long)gm * 1024 + col;
          if (which == 0)      qb[idx] = (__bf16)(val * qscale);
          else if (which == 1) kb[idx] = (__bf16)val;
          else                 vb[idx] = (__bf16)val;
        } else {
          C[(long long)gm * ldc + gn] = val;
        }
      }
    }
  }
}

// ---------------------------------------------------------------------------
// v [b][2048][1024] -> vt [b][1024][2048]  (bf16, 32x32 LDS tiles)
// ---------------------------------------------------------------------------
__global__ __launch_bounds__(256) void transpose_v_kernel(
    const __bf16* __restrict__ vin, __bf16* __restrict__ vout) {
  __shared__ __bf16 tile[32][33];
  const int b  = blockIdx.z;
  const int d0 = blockIdx.x * 32;
  const int t0 = blockIdx.y * 32;
  const int tx = threadIdx.x;  // 0..31
  const int ty = threadIdx.y;  // 0..7
  const __bf16* src = vin + (long long)b * 2048 * 1024;
  __bf16* dst       = vout + (long long)b * 1024 * 2048;
#pragma unroll
  for (int r = ty; r < 32; r += 8)
    tile[r][tx] = src[(long long)(t0 + r) * 1024 + d0 + tx];
  __syncthreads();
#pragma unroll
  for (int r = ty; r < 32; r += 8)
    dst[(long long)(d0 + r) * 2048 + t0 + tx] = tile[tx][r];
}

// ---------------------------------------------------------------------------
// Row softmax: 8192 rows x 2048 cols, f32 in -> bf16 out. One block per row.
// ---------------------------------------------------------------------------
__global__ __launch_bounds__(256) void softmax_rows_kernel(
    const float* __restrict__ S, __bf16* __restrict__ P) {
  __shared__ float red[256];
  const long long row = blockIdx.x;
  const float* s = S + row * 2048;
  __bf16* p      = P + row * 2048;
  const int tid = threadIdx.x;

  float v[8];
  float m = -3.0e38f;
#pragma unroll
  for (int i = 0; i < 8; ++i) { v[i] = s[tid + i * 256]; m = fmaxf(m, v[i]); }
  red[tid] = m; __syncthreads();
  for (int off = 128; off > 0; off >>= 1) {
    if (tid < off) red[tid] = fmaxf(red[tid], red[tid + off]);
    __syncthreads();
  }
  m = red[0]; __syncthreads();

  float sum = 0.0f;
#pragma unroll
  for (int i = 0; i < 8; ++i) { v[i] = __expf(v[i] - m); sum += v[i]; }
  red[tid] = sum; __syncthreads();
  for (int off = 128; off > 0; off >>= 1) {
    if (tid < off) red[tid] += red[tid + off];
    __syncthreads();
  }
  const float inv = 1.0f / red[0];
#pragma unroll
  for (int i = 0; i < 8; ++i) p[tid + i * 256] = (__bf16)(v[i] * inv);
}

// ---------------------------------------------------------------------------
// Launch: x[4,2048,1024] f32, W_qkv[3072,1024] f32 -> out[4,2048,1024] f32
// ---------------------------------------------------------------------------
extern "C" void kernel_launch(void* const* d_in, const int* in_sizes, int n_in,
                              void* d_out, int out_size, void* d_ws, size_t ws_size,
                              hipStream_t stream) {
  const float* x = (const float*)d_in[0];
  const float* W = (const float*)d_in[1];
  float* out     = (float*)d_out;
  char* ws       = (char*)d_ws;

  const size_t MB = 1024ull * 1024ull;
  // Lifetimes:  qb/kb: GEMM1->GEMM2   vb: GEMM1->transpose   vt: ->GEMM3
  //             xb/wb: pack->GEMM1    S: GEMM2->softmax (overlaps dead xb/wb)
  //             P: softmax->GEMM3 (overlaps dead qb/kb).  Peak = 128 MB.
  __bf16* qb = (__bf16*)(ws + 0);
  __bf16* kb = (__bf16*)(ws + 16 * MB);
  __bf16* vb = (__bf16*)(ws + 32 * MB);
  __bf16* vt = (__bf16*)(ws + 48 * MB);
  __bf16* xb = (__bf16*)(ws + 64 * MB);
  __bf16* wb = (__bf16*)(ws + 80 * MB);
  float*  S  = (float*)(ws + 64 * MB);   // 64 MB, written after xb/wb are dead
  __bf16* P  = (__bf16*)(ws + 0);        // 32 MB, written after qb/kb are dead

  const int nX = 4 * 2048 * 1024;   // 8388608
  const int nW = 3072 * 1024;       // 3145728

  f32_to_bf16_kernel<<<(nX + 255) / 256, 256, 0, stream>>>(x, xb, nX);
  f32_to_bf16_kernel<<<(nW + 255) / 256, 256, 0, stream>>>(W, wb, nW);

  // GEMM1: qkv = x @ W^T ; M=8192, N=3072, K=1024 ; de-interleave epilogue
  gemm_wmma_kernel<0><<<dim3(3072 / BN, 8192 / BM, 1), 256, 0, stream>>>(
      xb, 0, 1024, wb, 0, 1024, nullptr, 0, 0, 1024, 0.03125f, qb, kb, vb);

  transpose_v_kernel<<<dim3(1024 / 32, 2048 / 32, 4), dim3(32, 8), 0, stream>>>(vb, vt);

  // GEMM2 (per batch): S = q @ k^T ; M=N=2048, K=1024 ; q pre-scaled
  gemm_wmma_kernel<1><<<dim3(2048 / BN, 2048 / BM, 4), 256, 0, stream>>>(
      qb, 2048ll * 1024, 1024, kb, 2048ll * 1024, 1024,
      S, 2048ll * 2048, 2048, 1024, 1.0f, nullptr, nullptr, nullptr);

  softmax_rows_kernel<<<8192, 256, 0, stream>>>(S, P);

  // GEMM3 (per batch): out = P @ v ; A=P [2048x2048], Bt=vt [1024x2048], K=2048
  gemm_wmma_kernel<1><<<dim3(1024 / BN, 2048 / BM, 4), 256, 0, stream>>>(
      P, 2048ll * 2048, 2048, vt, 1024ll * 2048, 2048,
      out, 2048ll * 1024, 1024, 2048, 1.0f, nullptr, nullptr, nullptr);
}
